// GLoRIALoss_17617955848632
// MI455X (gfx1250) — compile-verified, matrix operand order checked by reading, not run on hardware
//
#include <hip/hip_runtime.h>
#include <hip/hip_bf16.h>
#include <math.h>

// ---------------- problem constants ----------------
#define B_  32
#define D_  768
#define S_  196          // H*W
#define SP_ 224          // S padded to multiple of 32 (K padding for matmul 2)
#define SM_ 208          // S padded to multiple of 16 (M padding for matmul 1)
#define L_  77
#define LP_ 80           // L padded to multiple of 16
#define TEMP_G 10.0f
#define TEMP1  4.0f
#define TEMP2  5.0f
#define TEMP3  10.0f
#define EPS_   1e-8f

typedef __attribute__((ext_vector_type(16))) __bf16 v16bf;
typedef __attribute__((ext_vector_type(8)))  __bf16 v8bf;
typedef __attribute__((ext_vector_type(8)))  float  v8f;

// Unconditional 16-bit A/B WMMA fragment load: per-lane = one matrix row
// (K-contiguous), two 8-element (16B) chunks per the CDNA5 16-bit 16x32 layout.
__device__ inline v16bf load_frag(const __bf16* c0, const __bf16* c1) {
    v8bf lo = *reinterpret_cast<const v8bf*>(c0);
    v8bf hi = *reinterpret_cast<const v8bf*>(c1);
    v16bf r;
#pragma unroll
    for (int e = 0; e < 8; ++e) { r[e] = lo[e]; r[e + 8] = hi[e]; }
    return r;
}

// ---------------- kernel 0: f32 -> bf16 staging (padded) + word norms ----------------
__global__ void glr_convert(const float* __restrict__ img_local,
                            const float* __restrict__ text_local,
                            __bf16* __restrict__ tlh,     // [B, LP_, D]  rows >= L_ zero
                            __bf16* __restrict__ ctxTh,   // [B, SM_, D]  rows >= S_ zero
                            __bf16* __restrict__ ctxh,    // [B, D, SP_]  cols >= S_ zero
                            float*  __restrict__ n1g)     // [B, L_]
{
    size_t tid = (size_t)blockIdx.x * blockDim.x + threadIdx.x;
    size_t nth = (size_t)gridDim.x * blockDim.x;

    // text_local [B, L, D] -> tlh [B, LP_, D] with zero pad rows
    const size_t NT = (size_t)B_ * LP_ * D_;
    for (size_t x = tid; x < NT; x += nth) {
        size_t i   = x / ((size_t)LP_ * D_);
        size_t rem = x % ((size_t)LP_ * D_);
        size_t l   = rem / D_;
        size_t d   = rem % D_;
        float v = (l < L_) ? text_local[(i * L_ + l) * D_ + d] : 0.f;
        tlh[x] = (__bf16)v;
    }

    // img_local [B, D, S] -> ctxh [B, D, SP_] (K-pad) and ctxTh [B, SM_, D] (M-pad)
    const size_t NC = (size_t)B_ * D_ * SP_;
    for (size_t x = tid; x < NC; x += nth) {
        size_t j   = x / ((size_t)D_ * SP_);
        size_t rem = x % ((size_t)D_ * SP_);
        size_t d   = rem / SP_;
        size_t s   = rem % SP_;
        float v = (s < S_) ? img_local[(j * D_ + d) * S_ + s] : 0.f;
        ctxh[x] = (__bf16)v;
    }
    const size_t NCT = (size_t)B_ * SM_ * D_;
    for (size_t x = tid; x < NCT; x += nth) {
        size_t j   = x / ((size_t)SM_ * D_);
        size_t rem = x % ((size_t)SM_ * D_);
        size_t s   = rem / D_;
        size_t d   = rem % D_;
        float v = (s < S_) ? img_local[(j * D_ + d) * S_ + s] : 0.f;
        ctxTh[x] = (__bf16)v;
    }

    const size_t NR = (size_t)B_ * L_;
    for (size_t x = tid; x < NR; x += nth) {
        const float* r = text_local + x * D_;
        float s = 0.f;
        for (int d = 0; d < D_; ++d) { float v = r[d]; s += v * v; }
        n1g[x] = sqrtf(s);
    }
}

// ---------------- kernel 1: global contrastive loss ----------------
__global__ void glr_global(const float* __restrict__ img_global,
                           const float* __restrict__ text_global,
                           float* __restrict__ lossg)
{
    __shared__ float ni[B_], nt[B_];
    __shared__ float gs[B_ * B_];
    __shared__ float acc[64];
    int tid = threadIdx.x;

    if (tid < 64) {
        int which = tid >> 5, r = tid & 31;
        const float* p = (which ? text_global : img_global) + (size_t)r * D_;
        float s = 0.f;
        for (int d = 0; d < D_; ++d) s += p[d] * p[d];
        float n = fmaxf(sqrtf(s), 1e-12f);
        if (which) nt[r] = n; else ni[r] = n;
    }
    __syncthreads();

    for (int p = tid; p < B_ * B_; p += blockDim.x) {
        int a = p >> 5, b = p & 31;
        const float* x = img_global  + (size_t)a * D_;
        const float* y = text_global + (size_t)b * D_;
        float s = 0.f;
        for (int d = 0; d < D_; ++d) s += x[d] * y[d];
        gs[p] = s / (ni[a] * nt[b]) * TEMP_G;
    }
    __syncthreads();

    if (tid < 64) {
        int which = tid >> 5, r = tid & 31;
        float m = -1e30f;
        for (int k = 0; k < B_; ++k) {
            float v = which ? gs[k * B_ + r] : gs[r * B_ + k];
            m = fmaxf(m, v);
        }
        float s = 0.f;
        for (int k = 0; k < B_; ++k) {
            float v = which ? gs[k * B_ + r] : gs[r * B_ + k];
            s += expf(v - m);
        }
        acc[tid] = (logf(s) + m) - gs[r * B_ + r];
    }
    __syncthreads();
    if (tid == 0) {
        float s = 0.f;
        for (int k = 0; k < 64; ++k) s += acc[k];
        lossg[0] = 0.5f * s / (float)B_;
    }
}

// ---------------- kernel 2: per (text i, image j) local similarity ----------------
// blockIdx.x = i*32 + j ; 256 threads = 8 waves (wave32)
__global__ void glr_local_pair(const __bf16* __restrict__ ctxTh,  // [B,SM_,D]
                               const __bf16* __restrict__ tlh,    // [B,LP_,D]
                               const __bf16* __restrict__ ctxh,   // [B,D,SP_]
                               const float*  __restrict__ text_local, // [B,L,D] f32
                               const int*    __restrict__ amask,  // [B,L]
                               const float*  __restrict__ n1g,    // [B,L]
                               float* __restrict__ simw)          // [B*B], sim[i][j]
{
    __shared__ float  logits[S_ * LP_];   // 62.7 KB : logits then a1, [s][l]
    __shared__ __bf16 a2h[LP_ * SP_];     // 35.8 KB : a2 as B^T rows [l][s]
    __shared__ float  w12[LP_], n2sq[LP_];
    __shared__ int    maski[LP_];

    const int pair = blockIdx.x;
    const int i = pair >> 5, j = pair & 31;
    const int tid  = threadIdx.x;
    const int lane = tid & 31;
    const int wave = tid >> 5;
    const int lrow = lane & 15;          // fragment row within tile
    const int khalf = (lane >> 4) * 8;   // K sub-chunk select

    if (tid < LP_) {
        maski[tid] = (tid < L_) ? amask[i * L_ + tid] : 0;
        w12[tid] = 0.f;
        n2sq[tid] = 0.f;
    }
    __syncthreads();

    // ---- Phase A: logits[s][l] = sum_d ctxT[j][s][d] * tl[i][l][d] ----
    // Each wave: one M-tile (s) x all 5 N-tiles (l); A frag reused 5x.
    {
        const __bf16* A  = ctxTh + (size_t)j * SM_ * D_;
        const __bf16* Bt = tlh   + (size_t)i * LP_ * D_;
        const __bf16* brow0 = Bt + (size_t)(0 * 16 + lrow) * D_;
        const __bf16* brow1 = Bt + (size_t)(1 * 16 + lrow) * D_;
        const __bf16* brow2 = Bt + (size_t)(2 * 16 + lrow) * D_;
        const __bf16* brow3 = Bt + (size_t)(3 * 16 + lrow) * D_;
        const __bf16* brow4 = Bt + (size_t)(4 * 16 + lrow) * D_;
        for (int mt = wave; mt < SM_ / 16; mt += 8) {   // 13 M-tiles
            const __bf16* arow = A + (size_t)(mt * 16 + lrow) * D_;
            v8f acc0 = {}, acc1 = {}, acc2 = {}, acc3 = {}, acc4 = {};
            for (int kk = 0; kk < D_ / 32; ++kk) {
                const int k0 = kk * 32;
                v16bf af = load_frag(arow + k0 + khalf, arow + k0 + 16 + khalf);
                v16bf b0 = load_frag(brow0 + k0 + khalf, brow0 + k0 + 16 + khalf);
                acc0 = __builtin_amdgcn_wmma_f32_16x16x32_bf16(false, af, false, b0, (short)0, acc0, false, false);
                v16bf b1 = load_frag(brow1 + k0 + khalf, brow1 + k0 + 16 + khalf);
                acc1 = __builtin_amdgcn_wmma_f32_16x16x32_bf16(false, af, false, b1, (short)0, acc1, false, false);
                v16bf b2 = load_frag(brow2 + k0 + khalf, brow2 + k0 + 16 + khalf);
                acc2 = __builtin_amdgcn_wmma_f32_16x16x32_bf16(false, af, false, b2, (short)0, acc2, false, false);
                v16bf b3 = load_frag(brow3 + k0 + khalf, brow3 + k0 + 16 + khalf);
                acc3 = __builtin_amdgcn_wmma_f32_16x16x32_bf16(false, af, false, b3, (short)0, acc3, false, false);
                v16bf b4 = load_frag(brow4 + k0 + khalf, brow4 + k0 + 16 + khalf);
                acc4 = __builtin_amdgcn_wmma_f32_16x16x32_bf16(false, af, false, b4, (short)0, acc4, false, false);
            }
            // C/D layout: lane = n + 16*(m>=8), vgpr r = m&7
            const int mbase = mt * 16 + (lane >> 4) * 8;
#pragma unroll
            for (int r = 0; r < 8; ++r) {
                const int s = mbase + r;
                if (s < S_) {
                    logits[s * LP_ +  0 + lrow] = acc0[r];
                    logits[s * LP_ + 16 + lrow] = acc1[r];
                    logits[s * LP_ + 32 + lrow] = acc2[r];
                    logits[s * LP_ + 48 + lrow] = acc3[r];
                    logits[s * LP_ + 64 + lrow] = acc4[r];
                }
            }
        }
    }
    __syncthreads();

    // ---- Phase B1: a1 = softmax over words l (masked) per pixel s ----
    if (tid < S_) {
        const int s = tid;
        float m = -1e30f;
        for (int l = 0; l < L_; ++l) if (maski[l]) m = fmaxf(m, logits[s * LP_ + l]);
        float sum = 0.f;
        for (int l = 0; l < L_; ++l) {
            float v = maski[l] ? expf(logits[s * LP_ + l] - m) : 0.f;
            logits[s * LP_ + l] = v;
            sum += v;
        }
        const float inv = 1.f / sum;
        for (int l = 0; l < L_; ++l) logits[s * LP_ + l] *= inv;
    }
    __syncthreads();

    // ---- Phase B2: a2[l][s] = softmax over pixels s of TEMP1*a1[s][l] ----
    if (tid < LP_) {
        const int l = tid;
        if (l < L_) {
            float m = -1e30f;
            for (int s = 0; s < S_; ++s) m = fmaxf(m, logits[s * LP_ + l]);
            m *= TEMP1;
            float sum = 0.f;
            for (int s = 0; s < S_; ++s) sum += expf(TEMP1 * logits[s * LP_ + l] - m);
            const float inv = 1.f / sum;
            for (int s = 0; s < S_; ++s)
                a2h[l * SP_ + s] = (__bf16)(expf(TEMP1 * logits[s * LP_ + l] - m) * inv);
            for (int s = S_; s < SP_; ++s) a2h[l * SP_ + s] = (__bf16)0.0f;
        } else {
            for (int s = 0; s < SP_; ++s) a2h[l * SP_ + s] = (__bf16)0.0f;
        }
    }
    __syncthreads();

    // ---- Phase C: wctx[d][l] = sum_s ctx[j][d][s]*a2[l][s]; fold into w12/n2sq ----
    // Each wave: one M-tile (d) x all 5 N-tiles (l); 48 M-tiles / 8 waves = 6 each.
    {
        const __bf16* A   = ctxh + (size_t)j * D_ * SP_;
        const float*  tli = text_local + (size_t)i * L_ * D_;   // [l][d]
        const __bf16* brow0 = &a2h[(0 * 16 + lrow) * SP_];
        const __bf16* brow1 = &a2h[(1 * 16 + lrow) * SP_];
        const __bf16* brow2 = &a2h[(2 * 16 + lrow) * SP_];
        const __bf16* brow3 = &a2h[(3 * 16 + lrow) * SP_];
        const __bf16* brow4 = &a2h[(4 * 16 + lrow) * SP_];
        for (int mt = wave; mt < D_ / 16; mt += 8) {
            const __bf16* arow = A + (size_t)(mt * 16 + lrow) * SP_;
            v8f acc0 = {}, acc1 = {}, acc2 = {}, acc3 = {}, acc4 = {};
#pragma unroll
            for (int kk = 0; kk < SP_ / 32; ++kk) {
                const int k0 = kk * 32;
                v16bf af = load_frag(arow + k0 + khalf, arow + k0 + 16 + khalf);
                v16bf b0 = load_frag(brow0 + k0 + khalf, brow0 + k0 + 16 + khalf);
                acc0 = __builtin_amdgcn_wmma_f32_16x16x32_bf16(false, af, false, b0, (short)0, acc0, false, false);
                v16bf b1 = load_frag(brow1 + k0 + khalf, brow1 + k0 + 16 + khalf);
                acc1 = __builtin_amdgcn_wmma_f32_16x16x32_bf16(false, af, false, b1, (short)0, acc1, false, false);
                v16bf b2 = load_frag(brow2 + k0 + khalf, brow2 + k0 + 16 + khalf);
                acc2 = __builtin_amdgcn_wmma_f32_16x16x32_bf16(false, af, false, b2, (short)0, acc2, false, false);
                v16bf b3 = load_frag(brow3 + k0 + khalf, brow3 + k0 + 16 + khalf);
                acc3 = __builtin_amdgcn_wmma_f32_16x16x32_bf16(false, af, false, b3, (short)0, acc3, false, false);
                v16bf b4 = load_frag(brow4 + k0 + khalf, brow4 + k0 + 16 + khalf);
                acc4 = __builtin_amdgcn_wmma_f32_16x16x32_bf16(false, af, false, b4, (short)0, acc4, false, false);
            }
            // Reduce all 5 tiles: lane column l = nt*16+lrow, rows d = mt*16 + (lane>>4)*8 + r
            const int dbase = mt * 16 + (lane >> 4) * 8;
#pragma unroll
            for (int nt = 0; nt < 5; ++nt) {
                const int l = nt * 16 + lrow;
                if (l < L_) {
                    const v8f* accp = (nt == 0) ? &acc0 : (nt == 1) ? &acc1 :
                                      (nt == 2) ? &acc2 : (nt == 3) ? &acc3 : &acc4;
                    float pw = 0.f, pn = 0.f;
#pragma unroll
                    for (int r = 0; r < 8; ++r) {
                        const float wv = (*accp)[r];
                        const float tv = tli[(size_t)l * D_ + dbase + r];
                        pw += tv * wv;
                        pn += wv * wv;
                    }
                    atomicAdd(&w12[l], pw);
                    atomicAdd(&n2sq[l], pn);
                }
            }
        }
    }
    __syncthreads();

    // ---- Phase D: sim[i][j] = log sum_l mask * exp(TEMP2 * cos_l) ----
    if (tid < LP_) {
        float val = 0.f;
        if (tid < L_ && maski[tid]) {
            const float n2 = sqrtf(n2sq[tid]);
            const float n1 = n1g[i * L_ + tid];
            const float cs = w12[tid] / fmaxf(n1 * n2, EPS_);
            val = expf(TEMP2 * cs);
        }
        w12[tid] = val;   // reuse as staging
    }
    __syncthreads();
    if (tid == 0) {
        float s = 0.f;
        for (int l = 0; l < L_; ++l) s += w12[l];
        simw[pair] = logf(s);
    }
}

// ---------------- kernel 3: local CE + combine ----------------
__global__ void glr_final(const float* __restrict__ simw,
                          const float* __restrict__ lossg,
                          float* __restrict__ out)
{
    __shared__ float acc[64];
    int tid = threadIdx.x;
    // scores[jj][ii] = simw[ii*32+jj] * TEMP3
    if (tid < 64) {
        int which = tid >> 5, r = tid & 31;
        float m = -1e30f;
        for (int k = 0; k < B_; ++k) {
            float v = (which ? simw[r * B_ + k] : simw[k * B_ + r]) * TEMP3;
            m = fmaxf(m, v);
        }
        float s = 0.f;
        for (int k = 0; k < B_; ++k) {
            float v = (which ? simw[r * B_ + k] : simw[k * B_ + r]) * TEMP3;
            s += expf(v - m);
        }
        acc[tid] = (logf(s) + m) - simw[r * B_ + r] * TEMP3;
    }
    __syncthreads();
    if (tid == 0) {
        float s = 0.f;
        for (int k = 0; k < 64; ++k) s += acc[k];
        out[0] = lossg[0] + 0.5f * s / (float)B_;
    }
}

// ---------------- host launch ----------------
extern "C" void kernel_launch(void* const* d_in, const int* in_sizes, int n_in,
                              void* d_out, int out_size, void* d_ws, size_t ws_size,
                              hipStream_t stream) {
    (void)in_sizes; (void)n_in; (void)out_size; (void)ws_size;
    const float* img_global  = (const float*)d_in[0];
    const float* img_local   = (const float*)d_in[1];
    const float* text_global = (const float*)d_in[2];
    const float* text_local  = (const float*)d_in[3];
    const int*   amask       = (const int*)  d_in[4];

    char* ws = (char*)d_ws;
    size_t off = 0;
    auto take = [&](size_t bytes) -> char* {
        char* p = ws + off;
        off = (off + bytes + 255) & ~(size_t)255;
        return p;
    };
    __bf16* tlh   = (__bf16*)take((size_t)B_ * LP_ * D_  * sizeof(__bf16));
    __bf16* ctxTh = (__bf16*)take((size_t)B_ * SM_ * D_  * sizeof(__bf16));
    __bf16* ctxh  = (__bf16*)take((size_t)B_ * D_  * SP_ * sizeof(__bf16));
    float*  n1g   = (float*) take((size_t)B_ * L_ * sizeof(float));
    float*  simw  = (float*) take((size_t)B_ * B_ * sizeof(float));
    float*  lossg = (float*) take(256);

    glr_convert   <<<1024, 256, 0, stream>>>(img_local, text_local, tlh, ctxTh, ctxh, n1g);
    glr_global    <<<1,    256, 0, stream>>>(img_global, text_global, lossg);
    glr_local_pair<<<B_ * B_, 256, 0, stream>>>(ctxTh, tlh, ctxh, text_local, amask, n1g, simw);
    glr_final     <<<1,     64, 0, stream>>>(simw, lossg, (float*)d_out);
}
